// Window_Hi_Attention_39204461478269
// MI455X (gfx1250) — matrix operand order repaired and checked by reading, not compiled, not run on hardware
//
#include <hip/hip_runtime.h>
#include <hip/hip_bf16.h>

// ---------------------------------------------------------------------------
// Window "high-head" attention for MI455X (gfx1250), wave32 + WMMA f16.
//   B=8192 windows, N=64 tokens, DIM=256, H_HEADS=4, HEAD_DIM=32, H_DIM=128
// ---------------------------------------------------------------------------

typedef _Float16 h16;
typedef __attribute__((ext_vector_type(16))) _Float16 v16h;
typedef __attribute__((ext_vector_type(8)))  _Float16 v8h;
typedef __attribute__((ext_vector_type(4)))  _Float16 v4h;
typedef __attribute__((ext_vector_type(8)))  float    v8f;
typedef __attribute__((ext_vector_type(4)))  float    v4f;

#define N_TOK   64
#define DIM     256
#define HH      4          // high heads
#define HD      32         // head dim
#define HDIM    128        // HH*HD
#define QKV_N   384        // 3*HDIM
#define SCALE_Q 0.17677669529663687f  // 32^-0.5

// LDS strides (halfs). All multiples of 8 so per-lane 8-half chunks are
// 16B aligned (ds_load_b128), and stride/2 dwords is odd*4 mod 64 -> no
// bank conflicts across the 16 lanes of a fragment half.
#define LDQ 136   // q / k buffers: 64 x 136
#define LDV 72    // vT buffer: 128 x 72
#define LDP 72    // x-chunk / P buffer: 64 x 72

static __device__ __forceinline__ v8f zero8() {
  v8f z = {0.f,0.f,0.f,0.f,0.f,0.f,0.f,0.f};
  return z;
}

// A-matrix fragment (16x32 f16): lane holds row (lane&15); 16 contiguous
// halfs at K = k0 + (lane>>4)*16.   (this code's k order, not the ISA table's; B uses the same)
static __device__ __forceinline__ v16h load_afrag(const h16* __restrict__ buf,
                                                  int ld, int r0, int k0, int lane) {
  const h16* p = buf + (r0 + (lane & 15)) * ld + k0 + ((lane >> 4) << 4);
  v8h lo = *(const v8h*)(p);
  v8h hi = *(const v8h*)(p + 8);
  v16h r;
#pragma unroll
  for (int i = 0; i < 8; ++i) { r[i] = lo[i]; r[i + 8] = hi[i]; }
  return r;
}

// B-matrix fragment (32x16 f16): lane holds column (lane&15); 16 contiguous
// halfs at K = k0 + (lane>>4)*16.  Requires K-major ("transposed") storage.
static __device__ __forceinline__ v16h load_bfrag(const h16* __restrict__ buf,
                                                  int ld, int c0, int k0, int lane) {
  const h16* p = buf + (c0 + (lane & 15)) * ld + k0 + ((lane >> 4) << 4);
  v8h lo = *(const v8h*)(p);
  v8h hi = *(const v8h*)(p + 8);
  v16h r;
#pragma unroll
  for (int i = 0; i < 8; ++i) { r[i] = lo[i]; r[i + 8] = hi[i]; }
  return r;
}

static __device__ __forceinline__ v8f wmma16(v16h a, v16h b, v8f c) {
  return __builtin_amdgcn_wmma_f32_16x16x32_f16(false, a, false, b,
                                                (short)0, c, false, false);
}

// ---------------------------------------------------------------------------
// Kernel 0: weight transpose to f16 (K-major for B-fragments) + bias build
// ---------------------------------------------------------------------------
__global__ void wha_prep(const float* __restrict__ qkv_w,   // [256][384]
                         const float* __restrict__ proj_w,  // [128][128]
                         const float* __restrict__ bias_table, // [225][4]
                         const int*   __restrict__ rel_index,  // [64][64]
                         h16* __restrict__ wT,      // [384][256]
                         h16* __restrict__ pT,      // [128][128]
                         float* __restrict__ biasws /* [4][64][64] */) {
  int i = blockIdx.x * blockDim.x + threadIdx.x;
  int stride = gridDim.x * blockDim.x;
  for (int t = i; t < QKV_N * DIM; t += stride) {
    int n = t >> 8, k = t & 255;
    wT[t] = (h16)qkv_w[k * QKV_N + n];
  }
  for (int t = i; t < HDIM * HDIM; t += stride) {
    int n = t >> 7, k = t & 127;
    pT[t] = (h16)proj_w[k * HDIM + n];
  }
  for (int t = i; t < HH * N_TOK * N_TOK; t += stride) {
    int h = t >> 12, nm = t & 4095;
    biasws[t] = bias_table[rel_index[nm] * HH + h];
  }
}

// ---------------------------------------------------------------------------
// Kernel 1: fused QKV GEMM + biased softmax attention.  One block per window.
// ---------------------------------------------------------------------------
__global__ __launch_bounds__(256) void wha_main(
    const float* __restrict__ x,        // [B][64][256]
    const float* __restrict__ qkv_b,    // [384]
    const h16*   __restrict__ wT,       // [384][256]
    const float* __restrict__ biasws,   // [4][64][64]
    h16*         __restrict__ Obuf)     // [B][64][128] f16
{
  __shared__ __align__(16) h16 sQ [N_TOK * LDQ];   // q * SCALE, [tok][128]
  __shared__ __align__(16) h16 sK [N_TOK * LDQ];   // k, [tok][128]
  __shared__ __align__(16) h16 sVT[HDIM  * LDV];   // v transposed, [feat][tok]
  __shared__ __align__(16) h16 sXP[N_TOK * LDP];   // x K-chunk, later P

  const int tid  = threadIdx.x;
  const int lane = tid & 31;
  const int w    = tid >> 5;          // wave 0..7
  const int b    = blockIdx.x;
  const int col   = lane & 15;        // fragment column / C-layout column
  const int rbase = (lane >> 4) * 8;  // C-layout row base

  const float* xb = x + (size_t)b * (N_TOK * DIM);

  // ---------------- QKV GEMM: (64x256) @ (256x384) ----------------
  // wave w owns row-tile mt = w>>1 and 12 column tiles nbase..nbase+11.
  v8f acc[12];
#pragma unroll
  for (int j = 0; j < 12; ++j) acc[j] = zero8();
  const int mt    = w >> 1;
  const int nbase = (w & 1) * 12;

  for (int kc = 0; kc < 4; ++kc) {       // K streamed in 4 chunks of 64
    __syncthreads();
    // stage x chunk -> f16 LDS (64 rows x 64 cols), float4 loads, v4h stores
#pragma unroll
    for (int i = 0; i < 4; ++i) {
      int j  = tid + i * 256;            // float4 index 0..1023
      int r  = j >> 4;
      int c4 = j & 15;
      v4f xv = *(const v4f*)(xb + r * DIM + kc * 64 + c4 * 4);
      v4h hv;
      hv[0] = (h16)xv[0]; hv[1] = (h16)xv[1];
      hv[2] = (h16)xv[2]; hv[3] = (h16)xv[3];
      *(v4h*)(&sXP[r * LDP + c4 * 4]) = hv;
    }
    __syncthreads();
#pragma unroll
    for (int kk = 0; kk < 2; ++kk) {     // two K=32 WMMA steps per chunk
      v16h a = load_afrag(sXP, LDP, mt * 16, kk * 32, lane);
#pragma unroll
      for (int j = 0; j < 12; ++j) {
        int nt = nbase + j;
        v16h bf = load_bfrag(wT, DIM, nt * 16, kc * 64 + kk * 32, lane);
        acc[j] = wmma16(a, bf, acc[j]);
      }
    }
  }

  // scatter q/k/vT into LDS (+bias; SCALE folded into q)
#pragma unroll
  for (int j = 0; j < 12; ++j) {
    int nt = nbase + j;
    int f  = nt * 16 + col;              // feature 0..383
    float bias = qkv_b[f];
    if (f < HDIM) {                      // q
#pragma unroll
      for (int e = 0; e < 8; ++e)
        sQ[(rbase + e) * LDQ + f] = (h16)((acc[j][e] + bias) * SCALE_Q);
    } else if (f < 2 * HDIM) {           // k
#pragma unroll
      for (int e = 0; e < 8; ++e)
        sK[(rbase + e) * LDQ + (f - HDIM)] = (h16)(acc[j][e] + bias);
    } else {                             // v, stored transposed [feat][tok]
#pragma unroll
      for (int e = 0; e < 8; ++e)
        sVT[(f - 2 * HDIM) * LDV + rbase + e] = (h16)(acc[j][e] + bias);
    }
  }
  __syncthreads();

  // ---------------- attention, head by head ----------------
  v8f oacc[4];
#pragma unroll
  for (int h = 0; h < HH; ++h) {
    if (w < 4) {                         // waves 0..3: S, softmax, P
      const int rt = w;
      v16h qa = load_afrag(sQ, LDQ, rt * 16, h * HD, lane);
      v8f s[4];
#pragma unroll
      for (int ct = 0; ct < 4; ++ct) {
        v16h kb = load_bfrag(sK, LDQ, ct * 16, h * HD, lane);
        s[ct] = wmma16(qa, kb, zero8());
      }
      const float* bh = biasws + h * (N_TOK * N_TOK);
      float mrow[8];
#pragma unroll
      for (int e = 0; e < 8; ++e) {
        int row = rt * 16 + rbase + e;
        float m = -3.0e38f;
#pragma unroll
        for (int ct = 0; ct < 4; ++ct) {
          s[ct][e] += bh[row * N_TOK + ct * 16 + col];
          m = fmaxf(m, s[ct][e]);
        }
        mrow[e] = m;
      }
#pragma unroll
      for (int e = 0; e < 8; ++e) {
        // row reduction across the 16 lanes of this half (rows differ by half)
        float m = mrow[e];
        m = fmaxf(m, __shfl_xor(m, 1, 32));
        m = fmaxf(m, __shfl_xor(m, 2, 32));
        m = fmaxf(m, __shfl_xor(m, 4, 32));
        m = fmaxf(m, __shfl_xor(m, 8, 32));
        float sum = 0.f;
#pragma unroll
        for (int ct = 0; ct < 4; ++ct) {
          s[ct][e] = __expf(s[ct][e] - m);
          sum += s[ct][e];
        }
        sum += __shfl_xor(sum, 1, 32);
        sum += __shfl_xor(sum, 2, 32);
        sum += __shfl_xor(sum, 4, 32);
        sum += __shfl_xor(sum, 8, 32);
        float rinv = 1.0f / sum;
        int row = rt * 16 + rbase + e;
#pragma unroll
        for (int ct = 0; ct < 4; ++ct)
          sXP[row * LDP + ct * 16 + col] = (h16)(s[ct][e] * rinv);
      }
    }
    __syncthreads();
    // P @ V : all 8 waves, one 16x16 O tile each (K = 64 -> 2 steps)
    {
      const int rt = w >> 1, dt = w & 1;
      v8f o = zero8();
#pragma unroll
      for (int ks = 0; ks < 2; ++ks) {
        v16h pa = load_afrag(sXP, LDP, rt * 16, ks * 32, lane);
        v16h vb = load_bfrag(sVT, LDV, h * HD + dt * 16, ks * 32, lane);
        o = wmma16(pa, vb, o);
      }
      oacc[h] = o;
    }
    __syncthreads();                      // sXP reused next head
  }

  // ---------------- write O (f16) to workspace ----------------
  h16* ob = Obuf + (size_t)b * (N_TOK * HDIM);
  const int rt = w >> 1, dt = w & 1;
#pragma unroll
  for (int h = 0; h < HH; ++h)
#pragma unroll
    for (int e = 0; e < 8; ++e)
      ob[(rt * 16 + rbase + e) * HDIM + h * HD + dt * 16 + col] =
          (h16)oacc[h][e];
}

// ---------------------------------------------------------------------------
// Kernel 2: output projection (64x128) @ (128x128) + bias, all from L2.
// ---------------------------------------------------------------------------
__global__ __launch_bounds__(256) void wha_proj(
    const h16*  __restrict__ Obuf,    // [B][64][128] f16
    const h16*  __restrict__ pT,      // [128][128] K-major
    const float* __restrict__ proj_b, // [128]
    float*       __restrict__ out)    // [B][64][128] f32
{
  const int tid  = threadIdx.x;
  const int lane = tid & 31;
  const int w    = tid >> 5;
  const int b    = blockIdx.x;
  const int col   = lane & 15;
  const int rbase = (lane >> 4) * 8;

  const h16* ob = Obuf + (size_t)b * (N_TOK * HDIM);
  const int mt = w >> 1, n0 = (w & 1) * 4;

  v8f acc[4];
#pragma unroll
  for (int j = 0; j < 4; ++j) acc[j] = zero8();

#pragma unroll
  for (int ks = 0; ks < 4; ++ks) {       // K = 128 -> 4 steps
    v16h a = load_afrag(ob, HDIM, mt * 16, ks * 32, lane);
#pragma unroll
    for (int j = 0; j < 4; ++j) {
      v16h bf = load_bfrag(pT, HDIM, (n0 + j) * 16, ks * 32, lane);
      acc[j] = wmma16(a, bf, acc[j]);
    }
  }

  float* outb = out + (size_t)b * (N_TOK * HDIM);
#pragma unroll
  for (int j = 0; j < 4; ++j) {
    int f = (n0 + j) * 16 + col;
    float pb = proj_b[f];
#pragma unroll
    for (int e = 0; e < 8; ++e)
      outb[(mt * 16 + rbase + e) * HDIM + f] = acc[j][e] + pb;
  }
}

// ---------------------------------------------------------------------------
extern "C" void kernel_launch(void* const* d_in, const int* in_sizes, int n_in,
                              void* d_out, int out_size, void* d_ws, size_t ws_size,
                              hipStream_t stream) {
  const float* x          = (const float*)d_in[0];
  const float* bias_table = (const float*)d_in[1];
  const float* qkv_w      = (const float*)d_in[2];
  const float* qkv_b      = (const float*)d_in[3];
  const float* proj_w     = (const float*)d_in[4];
  const float* proj_b     = (const float*)d_in[5];
  const int*   rel_index  = (const int*)d_in[6];

  const int B = in_sizes[0] / (N_TOK * DIM);   // 8192

  char* ws = (char*)d_ws;
  h16*   wT     = (h16*)(ws + 0);              // 384*256*2   = 196608
  h16*   pT     = (h16*)(ws + 196608);         // 128*128*2   =  32768
  float* biasws = (float*)(ws + 229376);       // 4*64*64*4   =  65536
  h16*   Obuf   = (h16*)(ws + 294912);         // B*64*128*2  = 128 MB
  float* out    = (float*)d_out;

  wha_prep<<<128, 256, 0, stream>>>(qkv_w, proj_w, bias_table, rel_index,
                                    wT, pT, biasws);
  wha_main<<<B, 256, 0, stream>>>(x, qkv_b, wT, biasws, Obuf);
  wha_proj<<<B, 256, 0, stream>>>(Obuf, pT, proj_b, out);
}
